// Bf16Module_15221364097544
// MI455X (gfx1250) — compile-verified, matrix operand order checked by reading, not run on hardware
//
#include <hip/hip_runtime.h>
#include <hip/hip_bf16.h>
#include <math.h>

// Problem dims (fixed by reference)
#define T_TOK 4096
#define D_EMB 1024
#define F_INN 2048
#define E_EXP 8

typedef __attribute__((ext_vector_type(16))) __bf16 v16bf;
typedef __attribute__((ext_vector_type(8)))  float  v8f;
typedef int v4i __attribute__((vector_size(16)));   // matches builtin param type

#if defined(__has_builtin)
#if __has_builtin(__builtin_amdgcn_global_load_async_to_lds_b128)
#define HAS_ASYNC_LDS 1
#endif
#endif
#ifndef HAS_ASYNC_LDS
#define HAS_ASYNC_LDS 0
#endif

#define AS1 __attribute__((address_space(1)))
#define AS3 __attribute__((address_space(3)))

union FragAB {
    uint4 q[2];
    v16bf v;
};

__device__ __forceinline__ unsigned short f2bf(float f) {
    unsigned u = __float_as_uint(f);
    unsigned r = u + 0x7FFFu + ((u >> 16) & 1u);   // round-to-nearest-even
    return (unsigned short)(r >> 16);
}

__device__ __forceinline__ float gelu_exact(float v) {
    return 0.5f * v * (1.0f + erff(v * 0.70710678118654752f));
}

// ---------------------------------------------------------------------------
// Zero the f32 output (harness poisons d_out with 0xAA)
// ---------------------------------------------------------------------------
__global__ void zero_f32_kernel(float4* __restrict__ p, long n4) {
    long i = (long)blockIdx.x * blockDim.x + threadIdx.x;
    if (i < n4) p[i] = make_float4(0.f, 0.f, 0.f, 0.f);
}

// ---------------------------------------------------------------------------
// Gating: logits = x @ wg^T, fp32 softmax, top-2 (no renorm).
// Writes dense combine weights transposed: cmbT[e*T + t].
// ---------------------------------------------------------------------------
__global__ void gate_topk_kernel(const float* __restrict__ x,
                                 const float* __restrict__ wg,
                                 float* __restrict__ cmbT) {
    int t = blockIdx.x * blockDim.x + threadIdx.x;
    if (t >= T_TOK) return;

    const float4* xr = reinterpret_cast<const float4*>(x + (size_t)t * D_EMB);
    float logit[E_EXP];
#pragma unroll
    for (int e = 0; e < E_EXP; ++e) {
        const float4* wr = reinterpret_cast<const float4*>(wg + (size_t)e * D_EMB);
        float acc = 0.f;
        for (int k = 0; k < D_EMB / 4; ++k) {
            float4 a = xr[k];
            float4 b = wr[k];
            acc = fmaf(a.x, b.x, acc);
            acc = fmaf(a.y, b.y, acc);
            acc = fmaf(a.z, b.z, acc);
            acc = fmaf(a.w, b.w, acc);
        }
        logit[e] = acc;
    }

    float m = logit[0];
#pragma unroll
    for (int e = 1; e < E_EXP; ++e) m = fmaxf(m, logit[e]);
    float p[E_EXP];
    float s = 0.f;
#pragma unroll
    for (int e = 0; e < E_EXP; ++e) { p[e] = expf(logit[e] - m); s += p[e]; }
    float inv = 1.0f / s;

    int i1 = 0;
#pragma unroll
    for (int e = 1; e < E_EXP; ++e) if (logit[e] > logit[i1]) i1 = e;
    int i2 = (i1 == 0) ? 1 : 0;
#pragma unroll
    for (int e = 0; e < E_EXP; ++e) if (e != i1 && logit[e] > logit[i2]) i2 = e;

#pragma unroll
    for (int e = 0; e < E_EXP; ++e)
        cmbT[(size_t)e * T_TOK + t] = (e == i1 || e == i2) ? p[e] * inv : 0.0f;
}

// ---------------------------------------------------------------------------
// f32 -> bf16 conversion, 2 elements / thread packed into one dword store
// ---------------------------------------------------------------------------
__global__ void cvt_f32_to_bf16_pairs(const float* __restrict__ src,
                                      unsigned short* __restrict__ dst,
                                      long npairs) {
    long i = (long)blockIdx.x * blockDim.x + threadIdx.x;
    if (i >= npairs) return;
    float2 v = reinterpret_cast<const float2*>(src)[i];
    unsigned u = (unsigned)f2bf(v.x) | ((unsigned)f2bf(v.y) << 16);
    reinterpret_cast<unsigned*>(dst)[i] = u;
}

// ---------------------------------------------------------------------------
// w2[E][F][D] f32 -> w2t[E][D][F] bf16 (LDS-tiled transpose, 64x64 tiles)
// Makes the fc2 B-operand K(=F)-contiguous for WMMA fragment loads.
// ---------------------------------------------------------------------------
__global__ __launch_bounds__(256)
void transpose_w2_kernel(const float* __restrict__ w2,
                         unsigned short* __restrict__ w2t) {
    __shared__ float tile[64][65];
    const int e  = blockIdx.z;
    const int d0 = blockIdx.x * 64;
    const int f0 = blockIdx.y * 64;

    const float* src = w2 + ((size_t)e * F_INN + f0) * D_EMB + d0;
    const int c4 = (threadIdx.x & 15) * 4;   // d within tile (float4)
    const int r  = threadIdx.x >> 4;         // f within tile, 16 rows/pass
#pragma unroll
    for (int i = 0; i < 4; ++i) {
        int row = r + i * 16;
        float4 v = *reinterpret_cast<const float4*>(src + (size_t)row * D_EMB + c4);
        tile[row][c4 + 0] = v.x;
        tile[row][c4 + 1] = v.y;
        tile[row][c4 + 2] = v.z;
        tile[row][c4 + 3] = v.w;
    }
    __syncthreads();

    unsigned short* dst = w2t + ((size_t)e * D_EMB + d0) * F_INN + f0;
    const int fc = (threadIdx.x & 31) * 2;   // f in output row (pairs)
    const int dr = threadIdx.x >> 5;         // 0..7
#pragma unroll
    for (int i = 0; i < 8; ++i) {
        int d = dr + i * 8;
        unsigned u = (unsigned)f2bf(tile[fc][d]) | ((unsigned)f2bf(tile[fc + 1][d]) << 16);
        *reinterpret_cast<unsigned*>(dst + (size_t)d * F_INN + fc) = u;
    }
}

// ---------------------------------------------------------------------------
// Tiled bf16 WMMA GEMM:  C[M,N] = A[M,K](row-major) * B[N,K](row-major)^T
//   MODE 0: OutBf[m,n]  = bf16(gelu_exact(acc))        (fc1 -> h)
//   MODE 1: OutF[m,n]  += scale[m] * acc               (fc2 -> y)
// 128x128 block tile, BK=64, 128 threads = 4 waves (2x2), 64x64 per wave:
//   16x v_wmma_f32_16x16x32_bf16 per 32-wide k-step, 1:1 with ds_load_b128.
// Double-buffered LDS staging via GLOBAL_LOAD_ASYNC_TO_LDS_B128 (ASYNCcnt),
// overlapping the next tile's DMA with the current tile's WMMAs.
// ---------------------------------------------------------------------------
template <int MODE>
__global__ __launch_bounds__(128)
void gemm_bf16_wmma(const unsigned short* __restrict__ A,
                    const unsigned short* __restrict__ B,
                    unsigned short* __restrict__ OutBf,
                    float* __restrict__ OutF,
                    const float* __restrict__ scale,
                    int M, int N, int K) {
    constexpr int BM  = 128, BN = 128, BK = 64;
    constexpr int LDT = BK + 8;  // padded LDS stride (bf16): 144B rows, conflict-free frags
    __shared__ alignas(16) unsigned short ldsA[2][BM * LDT];
    __shared__ alignas(16) unsigned short ldsB[2][BN * LDT];

    const int tid  = threadIdx.x;
    const int lane = tid & 31;
    const int wave = tid >> 5;
    const int wm   = wave >> 1;   // 0..1 : 64-row slice
    const int wn   = wave & 1;    // 0..1 : 64-col slice

    const int blockM = blockIdx.x * BM;
    const int blockN = blockIdx.y * BN;

    v8f acc[4][4] = {};

    // staging: 8 threads x 16B per 64-wide row; 16 rows/pass, 8 passes per matrix
    const int sr = tid >> 3;         // 0..15
    const int sc = (tid & 7) * 8;    // bf16 column offset (16B granules)

    auto stage = [&](int k0, int buf) {
        unsigned short* lA = &ldsA[buf][0];
        unsigned short* lB = &ldsB[buf][0];
#pragma unroll
        for (int i = 0; i < 8; ++i) {
            const int row = sr + i * 16;
            const unsigned short* ga = A + (size_t)(blockM + row) * K + k0 + sc;
            const unsigned short* gb = B + (size_t)(blockN + row) * K + k0 + sc;
            unsigned short* la = lA + row * LDT + sc;
            unsigned short* lb = lB + row * LDT + sc;
#if HAS_ASYNC_LDS
            __builtin_amdgcn_global_load_async_to_lds_b128(
                (AS1 v4i*)ga, (AS3 v4i*)la, 0, 0);
            __builtin_amdgcn_global_load_async_to_lds_b128(
                (AS1 v4i*)gb, (AS3 v4i*)lb, 0, 0);
#else
            *reinterpret_cast<uint4*>(la) = *reinterpret_cast<const uint4*>(ga);
            *reinterpret_cast<uint4*>(lb) = *reinterpret_cast<const uint4*>(gb);
#endif
        }
    };

    auto wait_stage = [&]() {
#if HAS_ASYNC_LDS
        asm volatile("s_wait_asynccnt 0" ::: "memory");
#endif
        __syncthreads();
    };

    // Prologue: fill buffer 0
    stage(0, 0);
    wait_stage();

    int buf = 0;
    for (int k0 = 0; k0 < K; k0 += BK) {
        if (k0 + BK < K) stage(k0 + BK, buf ^ 1);  // async DMA of next tile

#pragma unroll
        for (int kk = 0; kk < BK; kk += 32) {
            // A 16x32 bf16 frag: lane<16 -> K[fk..fk+7] in v0..3, K[fk+16..+23] in v4..7
            const int fr = lane & 15;
            const int fk = kk + ((lane >> 4) << 3);
            FragAB fa[4], fb[4];
#pragma unroll
            for (int mi = 0; mi < 4; ++mi) {
                const unsigned short* p = &ldsA[buf][(wm * 64 + mi * 16 + fr) * LDT + fk];
                fa[mi].q[0] = *reinterpret_cast<const uint4*>(p);
                fa[mi].q[1] = *reinterpret_cast<const uint4*>(p + 16);
            }
#pragma unroll
            for (int ni = 0; ni < 4; ++ni) {
                const unsigned short* p = &ldsB[buf][(wn * 64 + ni * 16 + fr) * LDT + fk];
                fb[ni].q[0] = *reinterpret_cast<const uint4*>(p);
                fb[ni].q[1] = *reinterpret_cast<const uint4*>(p + 16);
            }
#pragma unroll
            for (int mi = 0; mi < 4; ++mi)
#pragma unroll
                for (int ni = 0; ni < 4; ++ni)
                    acc[mi][ni] = __builtin_amdgcn_wmma_f32_16x16x32_bf16(
                        false, fa[mi].v, false, fb[ni].v,
                        (short)0, acc[mi][ni], false, false);
        }

        wait_stage();   // next buffer's DMA done + all waves finished reading `buf`
        buf ^= 1;
    }

    // Epilogue. C/D layout: lane<16 -> (M=r,    N=lane),
    //                       lane>=16 -> (M=r+8, N=lane-16), r = VGPR index 0..7
    const int colInLane = lane & 15;
    const int rowHalf   = (lane >> 4) << 3;  // 0 or 8
#pragma unroll
    for (int mi = 0; mi < 4; ++mi) {
#pragma unroll
        for (int ni = 0; ni < 4; ++ni) {
            const int col     = blockN + wn * 64 + ni * 16 + colInLane;
            const int rowBase = blockM + wm * 64 + mi * 16 + rowHalf;
#pragma unroll
            for (int r = 0; r < 8; ++r) {
                float v = acc[mi][ni][r];
                const size_t idx = (size_t)(rowBase + r) * N + col;
                if (MODE == 0) {
                    OutBf[idx] = f2bf(gelu_exact(v));
                } else {
                    OutF[idx] += scale[rowBase + r] * v;
                }
            }
        }
    }
}

// ---------------------------------------------------------------------------
// Launch: zero y, gate, convert to bf16, transpose w2, then per expert:
//   h = bf16(gelu(x @ w1e^T));  y += cmbT[e] * (h @ w2e)
// All launches on `stream`; scratch in d_ws (~92.4 MB).
// ---------------------------------------------------------------------------
extern "C" void kernel_launch(void* const* d_in, const int* in_sizes, int n_in,
                              void* d_out, int out_size, void* d_ws, size_t ws_size,
                              hipStream_t stream) {
    (void)in_sizes; (void)n_in; (void)out_size; (void)ws_size;

    const float* x  = (const float*)d_in[0];  // [T,D]
    const float* wg = (const float*)d_in[1];  // [E,D]
    const float* w1 = (const float*)d_in[2];  // [E,F,D]
    const float* w2 = (const float*)d_in[3];  // [E,F,D]
    float* y = (float*)d_out;                 // [T,D]

    char* ws = (char*)d_ws;
    float*          cmbT = (float*)(ws);                     //  131,072 B  [E][T]
    unsigned short* xbf  = (unsigned short*)(ws +   131072); // 8,388,608 B [T][D]
    unsigned short* hbf  = (unsigned short*)(ws +  8519680); // 16,777,216 B [T][F]
    unsigned short* w1bf = (unsigned short*)(ws + 25296896); // 33,554,432 B [E][F][D]
    unsigned short* w2t  = (unsigned short*)(ws + 58851328); // 33,554,432 B [E][D][F]

    zero_f32_kernel<<<(long)T_TOK * D_EMB / 4 / 256, 256, 0, stream>>>(
        (float4*)y, (long)T_TOK * D_EMB / 4);

    gate_topk_kernel<<<T_TOK / 256, 256, 0, stream>>>(x, wg, cmbT);

    cvt_f32_to_bf16_pairs<<<(long)T_TOK * D_EMB / 2 / 256, 256, 0, stream>>>(
        x, xbf, (long)T_TOK * D_EMB / 2);
    cvt_f32_to_bf16_pairs<<<(long)E_EXP * F_INN * D_EMB / 2 / 256, 256, 0, stream>>>(
        w1, w1bf, (long)E_EXP * F_INN * D_EMB / 2);
    transpose_w2_kernel<<<dim3(D_EMB / 64, F_INN / 64, E_EXP), 256, 0, stream>>>(
        w2, w2t);

    for (int e = 0; e < E_EXP; ++e) {
        gemm_bf16_wmma<0><<<dim3(T_TOK / 128, F_INN / 128), 128, 0, stream>>>(
            xbf, w1bf + (size_t)e * F_INN * D_EMB, hbf, nullptr, nullptr,
            T_TOK, F_INN, D_EMB);
        gemm_bf16_wmma<1><<<dim3(T_TOK / 128, D_EMB / 128), 128, 0, stream>>>(
            hbf, w2t + (size_t)e * D_EMB * F_INN, nullptr, y,
            cmbT + (size_t)e * T_TOK, T_TOK, D_EMB, F_INN);
    }
}
